// GTLayer_59639915872533
// MI455X (gfx1250) — matrix operand order repaired and checked by reading, not compiled, not run on hardware
//
#include <hip/hip_runtime.h>

// ---------------------------------------------------------------------------
// CDNA5 (gfx1250, wave32) graph-transformer layer.
// GEMMs: V_WMMA_F32_16X16X32_BF16; BOTH operand tiles staged by the Tensor
// Data Mover (tensor_load_to_lds) with double buffering. Weights are
// pre-transposed to make B tiles K-contiguous for the TDM.
// ---------------------------------------------------------------------------

typedef __attribute__((ext_vector_type(16))) __bf16 v16bf;
typedef __attribute__((ext_vector_type(8)))  float  v8f;
typedef __attribute__((ext_vector_type(4)))  unsigned v4u;
typedef __attribute__((ext_vector_type(8)))  int      v8i;
typedef __attribute__((ext_vector_type(4)))  int      v4i;

union BF16Frag { v16bf v; unsigned u[8]; unsigned short s[16]; };

__device__ __forceinline__ unsigned short f2bf(float f) {
    unsigned u = __float_as_uint(f);
    u += 0x7FFFu + ((u >> 16) & 1u);          // round-to-nearest-even
    return (unsigned short)(u >> 16);
}
__device__ __forceinline__ unsigned flipf(float f) {      // order-preserving f32->u32
    unsigned u = __float_as_uint(f);
    return (u & 0x80000000u) ? ~u : (u | 0x80000000u);
}
__device__ __forceinline__ float unflipf(unsigned e) {
    unsigned u = (e & 0x80000000u) ? (e & 0x7FFFFFFFu) : ~e;
    return __uint_as_float(u);
}

// ---------------------------------------------------------------------------
// Utility kernels
// ---------------------------------------------------------------------------
__global__ void f2bf_kernel(const float* __restrict__ in,
                            unsigned short* __restrict__ out, long n) {
    long i = (long)blockIdx.x * blockDim.x + threadIdx.x;
    if (i < n) out[i] = f2bf(in[i]);
}

// out[n*K + k] = bf16(in[k*N + n])  (weights are small; one-time cost)
__global__ void f2bf_transpose_kernel(const float* __restrict__ in,
                                      unsigned short* __restrict__ out,
                                      int K, int N) {
    long i = (long)blockIdx.x * blockDim.x + threadIdx.x;
    if (i < (long)K * N) {
        int k = (int)(i / N), n = (int)(i % N);
        out[(size_t)n * K + k] = f2bf(in[i]);
    }
}

__global__ void fill_u32_kernel(unsigned* __restrict__ p, unsigned v, long n) {
    long i = (long)blockIdx.x * blockDim.x + threadIdx.x;
    if (i < n) p[i] = v;
}

// ---------------------------------------------------------------------------
// TDM issue helper: 2D tile of 16-bit data, 32 contiguous K per row, `rows`
// rows, source row stride = strideK elements; pad LDS dest +1 DWORD every
// 16 DWORDs (=> 34-ushort LDS row stride); OOB rows/cols hardware-zero-filled.
// ---------------------------------------------------------------------------
__device__ __forceinline__ void tdm_load_tile(unsigned ldsAddr,
                                              const unsigned short* gptr,
                                              unsigned dim0, unsigned dim1,
                                              unsigned rows, unsigned strideK) {
    unsigned long long gaddr = (unsigned long long)(uintptr_t)gptr;
    v4u g0;
    g0[0] = 1u;                                   // count=1, user descriptor
    g0[1] = ldsAddr;                              // lds_addr
    g0[2] = (unsigned)(gaddr & 0xFFFFFFFFu);      // global_addr[31:0]
    g0[3] = (unsigned)((gaddr >> 32) & 0x01FFFFFFu) | (2u << 30); // addr[56:32], type=2
    v8i g1;
    g1[0] = (int)((1u << 16) |                    // data_size = 2 bytes
                  (1u << 20) |                    // pad_enable
                  (3u << 22));                    // pad_interval = 16 DWORDs (pad_amount=0 -> 1 DWORD)
    g1[1] = (int)((dim0 & 0xFFFFu) << 16);        // tensor_dim0[15:0] (atomic barrier addr = 0)
    g1[2] = (int)((dim0 >> 16) | ((dim1 & 0xFFFFu) << 16));  // dim0 hi | dim1 lo
    g1[3] = (int)((dim1 >> 16) | (32u << 16));    // dim1 hi | tile_dim0 = 32
    g1[4] = (int)rows;                            // tile_dim1 = rows, tile_dim2 = 0
    g1[5] = (int)strideK;                         // tensor_dim0_stride lo
    g1[6] = 0;
    g1[7] = 0;
    v4i gz4 = {0, 0, 0, 0};
    v8i gz8 = {0, 0, 0, 0, 0, 0, 0, 0};
    __builtin_amdgcn_tensor_load_to_lds(g0, g1, gz4, gz4, gz8, 0);
}

// ---------------------------------------------------------------------------
// WMMA bf16 GEMM:  C[M,Nn] = A[M,K] @ B[K,Nn] + bias, optional relu.
// A row-major bf16 [M][K]; BT row-major bf16 [Nn][K] (pre-transposed B).
// Block tile 128x64, 8 waves, each wave 32x32 (2x2 WMMA), K-step 32.
// Both tiles DMA'd by the TDM, double buffered (issue next before waiting
// on current; TENSORcnt completes in order so wait<=2 == current pair done).
// ---------------------------------------------------------------------------
__global__ __launch_bounds__(256)
void wmma_gemm_bf16(const unsigned short* __restrict__ A,
                    const unsigned short* __restrict__ BT,
                    const float* __restrict__ bias,
                    float* __restrict__ outF,
                    unsigned short* __restrict__ outB,
                    int M, int Nn, int K, int relu) {
    constexpr int LDT = 34;                       // padded LDS stride (ushorts)
    __shared__ __align__(16) unsigned short As[2][128 * LDT];  // [row][k]
    __shared__ __align__(16) unsigned short Bs[2][64  * LDT];  // [col][k]

    const int tid   = threadIdx.x;
    const int lane  = tid & 31;
    const int wave  = tid >> 5;
    const int waveM = wave & 3;                   // 4 waves along M
    const int waveN = wave >> 2;                  // 2 waves along N
    const int blockM = blockIdx.x * 128;
    const int blockN = blockIdx.y * 64;
    const int half = lane >> 4;
    const int l15  = lane & 15;

    v8f acc[2][2] = {};

    // prologue: stage k0=0 into buffer 0
    if (wave == 0) {
        tdm_load_tile((unsigned)(uintptr_t)(&As[0][0]),
                      A + (size_t)blockM * K,
                      (unsigned)K, (unsigned)(M - blockM), 128u, (unsigned)K);
        tdm_load_tile((unsigned)(uintptr_t)(&Bs[0][0]),
                      BT + (size_t)blockN * K,
                      (unsigned)K, (unsigned)(Nn - blockN), 64u, (unsigned)K);
    }

    for (int k0 = 0; k0 < K; k0 += 32) {
        const int cur = (k0 >> 5) & 1;
        const int nxt = cur ^ 1;
        const bool more = (k0 + 32) < K;

        if (wave == 0) {
            if (more) {                            // prefetch next pair
                tdm_load_tile((unsigned)(uintptr_t)(&As[nxt][0]),
                              A + (size_t)blockM * K + (k0 + 32),
                              (unsigned)(K - (k0 + 32)), (unsigned)(M - blockM),
                              128u, (unsigned)K);
                tdm_load_tile((unsigned)(uintptr_t)(&Bs[nxt][0]),
                              BT + (size_t)blockN * K + (k0 + 32),
                              (unsigned)(K - (k0 + 32)), (unsigned)(Nn - blockN),
                              64u, (unsigned)K);
                __builtin_amdgcn_s_wait_tensorcnt(2);   // current pair landed
            } else {
                __builtin_amdgcn_s_wait_tensorcnt(0);   // last pair landed
            }
        }
        __syncthreads();

        const unsigned short* Ac = &As[cur][0];
        const unsigned short* Bc = &Bs[cur][0];

        // ---- B fragments (16-bit B layout: lane N=l15, same K order as A) ----
        BF16Frag bf[2];
        #pragma unroll
        for (int tn = 0; tn < 2; ++tn) {
            int col = waveN * 32 + tn * 16 + l15;
            #pragma unroll
            for (int j = 0; j < 8; ++j) {
                int kk = ((j < 4) ? (2 * j) : (16 + 2 * (j - 4))) + 8 * half;
                bf[tn].u[j] = *(const unsigned*)(&Bc[col * LDT + kk]);
            }
        }
        // ---- A fragments (ISA 16-bit A 16x32 layout) + WMMA ----
        #pragma unroll
        for (int tm = 0; tm < 2; ++tm) {
            int row = waveM * 32 + tm * 16 + l15;
            BF16Frag af;
            #pragma unroll
            for (int j = 0; j < 8; ++j) {
                int kk = ((j < 4) ? (2 * j) : (16 + 2 * (j - 4))) + 8 * half;
                af.u[j] = *(const unsigned*)(&Ac[row * LDT + kk]);
            }
            #pragma unroll
            for (int tn = 0; tn < 2; ++tn) {
                acc[tm][tn] = __builtin_amdgcn_wmma_f32_16x16x32_bf16(
                    false, af.v, false, bf[tn].v, (short)0, acc[tm][tn], false, false);
            }
        }
        __syncthreads();   // all waves done with `cur` before it is re-staged
    }

    // ---- epilogue (C layout: elem j -> row j+8*half, col l15) ----
    #pragma unroll
    for (int tm = 0; tm < 2; ++tm) {
        #pragma unroll
        for (int tn = 0; tn < 2; ++tn) {
            int col = blockN + waveN * 32 + tn * 16 + l15;
            float bv = bias ? bias[col] : 0.0f;
            #pragma unroll
            for (int j = 0; j < 8; ++j) {
                int rowg = blockM + waveM * 32 + tm * 16 + half * 8 + j;
                if (rowg < M) {
                    float v = acc[tm][tn][j] + bv;
                    if (relu) v = v > 0.0f ? v : 0.0f;
                    if (outF) outF[(size_t)rowg * Nn + col] = v;
                    if (outB) outB[(size_t)rowg * Nn + col] = f2bf(v);
                }
            }
        }
    }
}

// ---------------------------------------------------------------------------
// Edge attention kernels.  qkv layout per node: [h*96 + {q:0..31, k:32..63, v:64..95}]
// ---------------------------------------------------------------------------
__global__ __launch_bounds__(256)
void edge_scores_kernel(const float* __restrict__ qkv,
                        const int* __restrict__ rows, const int* __restrict__ cols,
                        float* __restrict__ alpha, unsigned* __restrict__ menc, int E) {
    int wid  = (blockIdx.x * 256 + threadIdx.x) >> 5;   // one wave32 per edge
    int lane = threadIdx.x & 31;
    if (wid >= E) return;
    int r = rows[wid], c = cols[wid];
    const float* qb = qkv + (size_t)r * 768;
    const float* kb = qkv + (size_t)c * 768 + 32;
    #pragma unroll
    for (int h = 0; h < 8; ++h) {
        float p = qb[h * 96 + lane] * kb[h * 96 + lane];
        #pragma unroll
        for (int off = 16; off > 0; off >>= 1) p += __shfl_xor(p, off, 32);
        if (lane == 0) {
            float a = p * 0.0625f;                       // D^-0.5, D=256
            alpha[(size_t)wid * 8 + h] = a;
            atomicMax(&menc[r * 8 + h], flipf(a));
        }
    }
}

__global__ void edge_exp_kernel(const int* __restrict__ rows,
                                const unsigned* __restrict__ menc,
                                float* __restrict__ alpha,
                                float* __restrict__ denom, int E) {
    long idx = (long)blockIdx.x * blockDim.x + threadIdx.x;
    if (idx >= (long)E * 8) return;
    int e = (int)(idx >> 3), h = (int)(idx & 7);
    int r = rows[e];
    float m  = unflipf(menc[r * 8 + h]);
    float ea = expf(alpha[idx] - m);
    alpha[idx] = ea;
    atomicAdd(&denom[r * 8 + h], ea);
}

__global__ __launch_bounds__(256)
void edge_agg_kernel(const float* __restrict__ qkv,
                     const int* __restrict__ rows, const int* __restrict__ cols,
                     const float* __restrict__ ea, const float* __restrict__ denom,
                     float* __restrict__ attnout, int E) {
    int wid  = (blockIdx.x * 256 + threadIdx.x) >> 5;
    int lane = threadIdx.x & 31;
    if (wid >= E) return;
    int r = rows[wid], c = cols[wid];
    const float* vb = qkv + (size_t)c * 768 + 64;
    float* ob = attnout + (size_t)r * 256;
    #pragma unroll
    for (int h = 0; h < 8; ++h) {
        float w = ea[(size_t)wid * 8 + h] / denom[r * 8 + h];
        atomicAdd(&ob[h * 32 + lane], w * vb[h * 96 + lane]);
    }
}

// ---------------------------------------------------------------------------
// Residual + LayerNorm over D=256 (one block per row), optional bf16 copy out.
// ---------------------------------------------------------------------------
__global__ __launch_bounds__(256)
void residual_ln_kernel(const float* __restrict__ base, const float* __restrict__ add,
                        const float* __restrict__ g, const float* __restrict__ b,
                        float* __restrict__ outF, unsigned short* __restrict__ outB,
                        int Nrows) {
    int n = blockIdx.x;
    if (n >= Nrows) return;
    int t = threadIdx.x;
    __shared__ float red[256];
    size_t off = (size_t)n * 256 + t;
    float val = base[off] + add[off];
    red[t] = val; __syncthreads();
    for (int s = 128; s > 0; s >>= 1) { if (t < s) red[t] += red[t + s]; __syncthreads(); }
    float mean = red[0] * (1.0f / 256.0f);
    __syncthreads();
    float dv = val - mean;
    red[t] = dv * dv; __syncthreads();
    for (int s = 128; s > 0; s >>= 1) { if (t < s) red[t] += red[t + s]; __syncthreads(); }
    float var = red[0] * (1.0f / 256.0f);
    float y = dv * rsqrtf(var + 1e-5f) * g[t] + b[t];
    if (outF) outF[off] = y;
    if (outB) outB[off] = f2bf(y);
}

// ---------------------------------------------------------------------------
// Launcher
// ---------------------------------------------------------------------------
extern "C" void kernel_launch(void* const* d_in, const int* in_sizes, int n_in,
                              void* d_out, int out_size, void* d_ws, size_t ws_size,
                              hipStream_t stream) {
    (void)n_in; (void)out_size; (void)ws_size;
    const float* x     = (const float*)d_in[0];
    const int*   ei    = (const int*)  d_in[1];
    const float* w_qkv = (const float*)d_in[2];
    const float* b_qkv = (const float*)d_in[3];
    const float* ln1g  = (const float*)d_in[4];
    const float* ln1b  = (const float*)d_in[5];
    const float* ln2g  = (const float*)d_in[6];
    const float* ln2b  = (const float*)d_in[7];
    const float* w1    = (const float*)d_in[8];
    const float* b1    = (const float*)d_in[9];
    const float* w2    = (const float*)d_in[10];
    const float* b2    = (const float*)d_in[11];

    const int Nn = in_sizes[0] / 256;   // 50000
    const int E  = in_sizes[1] / 2;     // 800000
    const int* rows = ei;
    const int* cols = ei + E;

    char* ws = (char*)d_ws;
    size_t o = 0;
    auto alloc = [&](size_t bytes) -> char* {
        char* p = ws + o; o += (bytes + 255) & ~(size_t)255; return p;
    };
    unsigned short* x_bf   = (unsigned short*)alloc((size_t)Nn * 256 * 2);
    unsigned short* wqkvT  = (unsigned short*)alloc((size_t)768 * 256 * 2);   // [768][256]
    unsigned short* w1T    = (unsigned short*)alloc((size_t)1024 * 256 * 2);  // [1024][256]
    unsigned short* w2T    = (unsigned short*)alloc((size_t)256 * 1024 * 2);  // [256][1024]
    float*    qkv     = (float*)   alloc((size_t)Nn * 768 * 4);
    float*    alpha   = (float*)   alloc((size_t)E * 8 * 4);
    unsigned* menc    = (unsigned*)alloc((size_t)Nn * 8 * 4);
    float*    denom   = (float*)   alloc((size_t)Nn * 8 * 4);
    float*    attnout = (float*)   alloc((size_t)Nn * 256 * 4);
    float*    y1f     = (float*)   alloc((size_t)Nn * 256 * 4);
    unsigned short* y1_bf = (unsigned short*)alloc((size_t)Nn * 256 * 2);
    // reuse the qkv region (Nn*3072 B) once attention is done:
    unsigned short* h_bf = (unsigned short*)qkv;                 // Nn*1024 bf16
    float* ffn = (float*)((char*)qkv + (size_t)Nn * 1024 * 2);   // Nn*256 f32

    // --- bf16 conversions (weights transposed so B tiles are K-contiguous) ---
    long nx = (long)Nn * 256;
    f2bf_kernel<<<(int)((nx + 255) / 256), 256, 0, stream>>>(x, x_bf, nx);
    f2bf_transpose_kernel<<<(196608 + 255) / 256, 256, 0, stream>>>(w_qkv, wqkvT, 256, 768);
    f2bf_transpose_kernel<<<(262144 + 255) / 256, 256, 0, stream>>>(w1, w1T, 256, 1024);
    f2bf_transpose_kernel<<<(262144 + 255) / 256, 256, 0, stream>>>(w2, w2T, 1024, 256);

    // --- init accumulators (every call: graphs replay without re-poison) ---
    long nm = (long)Nn * 8;
    fill_u32_kernel<<<(int)((nm + 255) / 256), 256, 0, stream>>>(menc, 0u, nm);
    fill_u32_kernel<<<(int)((nm + 255) / 256), 256, 0, stream>>>((unsigned*)denom, 0u, nm);
    fill_u32_kernel<<<(int)((nx + 255) / 256), 256, 0, stream>>>((unsigned*)attnout, 0u, nx);

    // --- QKV GEMM: [Nn,256]@[256,768] + b_qkv ---
    dim3 g1((Nn + 127) / 128, 768 / 64);
    wmma_gemm_bf16<<<g1, 256, 0, stream>>>(x_bf, wqkvT, b_qkv, qkv, nullptr,
                                           Nn, 768, 256, 0);

    // --- edge softmax attention ---
    int eblocks = (E + 7) / 8;   // 8 wave32 edges per 256-thread block
    edge_scores_kernel<<<eblocks, 256, 0, stream>>>(qkv, rows, cols, alpha, menc, E);
    long eh = (long)E * 8;
    edge_exp_kernel<<<(int)((eh + 255) / 256), 256, 0, stream>>>(rows, menc, alpha, denom, E);
    edge_agg_kernel<<<eblocks, 256, 0, stream>>>(qkv, rows, cols, alpha, denom, attnout, E);

    // --- residual + LN1 ---
    residual_ln_kernel<<<Nn, 256, 0, stream>>>(x, attnout, ln1g, ln1b, y1f, y1_bf, Nn);

    // --- FFN ---
    dim3 g2((Nn + 127) / 128, 1024 / 64);
    wmma_gemm_bf16<<<g2, 256, 0, stream>>>(y1_bf, w1T, b1, nullptr, h_bf,
                                           Nn, 1024, 256, 1);
    dim3 g3((Nn + 127) / 128, 256 / 64);
    wmma_gemm_bf16<<<g3, 256, 0, stream>>>(h_bf, w2T, b2, ffn, nullptr,
                                           Nn, 256, 1024, 0);

    // --- residual + LN2 -> output ---
    residual_ln_kernel<<<Nn, 256, 0, stream>>>(y1f, ffn, ln2g, ln2b,
                                               (float*)d_out, nullptr, Nn);
}
